// PairwiseGPMapLayer_15264313770445
// MI455X (gfx1250) — matrix operand order repaired and checked by reading, not compiled, not run on hardware
//
#include <hip/hip_runtime.h>
#include <hip/hip_bf16.h>
#include <stddef.h>

// Problem constants (match reference)
#define LL      256
#define CC      20
#define BB      4096
#define LC      5120        // L*C
#define KSTEPS  160         // LC / 32   (K-dim WMMA steps)
#define NBLK    320         // LC / 16   (N 16-wide blocks)
#define MBLK    256         // B  / 16   (M 16-wide blocks)
#define NCHUNKS 80          // LC / 64   (N 64-wide chunks; partial-buffer granularity)

typedef __bf16 bf16x16 __attribute__((ext_vector_type(16)));
typedef float  f32x8   __attribute__((ext_vector_type(8)));

__device__ __forceinline__ unsigned short f32_to_bf16_rne(float f) {
  unsigned int u = __float_as_uint(f);
  u += 0x7FFFu + ((u >> 16) & 1u);   // round-to-nearest-even
  return (unsigned short)(u >> 16);
}

// ---------------------------------------------------------------------------
// Swizzle Theta (fp32 KxN row-major, K=N=5120) -> bf16 B-fragments.
// B 32x16 bf16 fragment layout (this code's contiguous K order): lane = (K>=16)*16 + N%16,
// element e in lane's 16 bf16 = K%16.  Fragment id = kStep*NBLK + nBlock,
// stored lane-major: frag[lane*16 + e].
// ---------------------------------------------------------------------------
__global__ void __launch_bounds__(256) pgm_swizzle_theta(
    const float* __restrict__ theta, unsigned short* __restrict__ dst) {
  size_t t = (size_t)blockIdx.x * 256 + threadIdx.x;
  if (t >= (size_t)LC * LC) return;
  int k = (int)(t / LC);
  int n = (int)(t % LC);
  int kStep = k >> 5, kIn = k & 31;
  int hi = kIn >> 4, e = kIn & 15;
  int nBlock = n >> 4, nl = n & 15;
  int lane = (hi << 4) | nl;
  size_t off = ((size_t)(kStep * NBLK + nBlock) * 32 + lane) * 16 + e;
  dst[off] = f32_to_bf16_rne(theta[t]);
}

// ---------------------------------------------------------------------------
// Swizzle X (fp32 BxK row-major) -> bf16 A-fragments.
// A 16x32 bf16 fragment layout (this code's contiguous K order): lane = laneHi*16 + M%16 with
// lanes 0-15 holding K = {0..15}, lanes 16-31 K = {16..31};
// element e: K = e + laneHi*16.  Frag id = mBlock*KSTEPS+kStep.
// ---------------------------------------------------------------------------
__global__ void __launch_bounds__(256) pgm_swizzle_x(
    const float* __restrict__ x, unsigned short* __restrict__ dst) {
  size_t t = (size_t)blockIdx.x * 256 + threadIdx.x;
  if (t >= (size_t)BB * LC) return;
  int b = (int)(t / LC);
  int j = (int)(t % LC);
  int mBlock = b >> 4, ml = b & 15;
  int kStep = j >> 5, kIn = j & 31;
  int laneHi = kIn >> 4;
  int e = kIn & 15;
  int lane = (laneHi << 4) | ml;
  size_t off = ((size_t)(mBlock * KSTEPS + kStep) * 32 + lane) * 16 + e;
  dst[off] = f32_to_bf16_rne(x[t]);
}

// ---------------------------------------------------------------------------
// Fused GEMM + elementwise-x + row reduction.
// Block = 256 threads = 8 waves arranged 4(M) x 2(N).  Each wave computes a
// 64(M) x 64(N) tile: 4 A-fragments x 4 B-fragments -> 16 independent WMMA
// accumulation chains per K-step (8 b128-pair loads : 16 WMMAs, 0.5 KB of
// cache traffic per WMMA).  Within the block, each A frag is read by 2
// waves and each B frag by 4 waves (WGP$ reuse); block tile = 256 x 128.
// Epilogue multiplies by fp32 x at each output position and reduces each
// row over the wave's 64 columns into partial[row*NCHUNKS + chunk]
// (exactly one deterministic writer per slot).
// ---------------------------------------------------------------------------
__global__ void __launch_bounds__(256) pgm_gemm(
    const unsigned short* __restrict__ aFrags,
    const unsigned short* __restrict__ bFrags,
    const float* __restrict__ x,
    float* __restrict__ partial) {
  const int lane = threadIdx.x & 31;
  const int w    = threadIdx.x >> 5;
  const int wm   = w & 3;                      // wave's M position (0..3)
  const int wn   = w >> 2;                     // wave's N position (0..1)
  const int mB0  = blockIdx.y * 16 + wm * 4;   // first of 4 mBlocks (0..252)
  const int nB0  = blockIdx.x * 8 + wn * 4;    // first of 4 nBlocks (0..316)

  f32x8 c[4][4] = {};

  const unsigned short* aBase = aFrags + ((size_t)mB0 * KSTEPS) * 512 + (size_t)lane * 16;
  const unsigned short* bBase = bFrags + ((size_t)nB0) * 512 + (size_t)lane * 16;

  for (int ks = 0; ks < KSTEPS; ++ks) {
    const unsigned short* ak = aBase + (size_t)ks * 512;
    const unsigned short* bk = bBase + (size_t)ks * (size_t)NBLK * 512;
    bf16x16 a[4], b[4];
#pragma unroll
    for (int m = 0; m < 4; ++m)
      a[m] = *(const bf16x16*)(ak + (size_t)m * KSTEPS * 512);
#pragma unroll
    for (int n = 0; n < 4; ++n)
      b[n] = *(const bf16x16*)(bk + (size_t)n * 512);
    if (ks + 1 < KSTEPS) {
      __builtin_prefetch(bk + (size_t)NBLK * 512, 0, 3);  // next K-step B stream
      __builtin_prefetch(ak + 512, 0, 3);                 // next K-step A stream
    }
#pragma unroll
    for (int m = 0; m < 4; ++m) {
#pragma unroll
      for (int n = 0; n < 4; ++n) {
        c[m][n] = __builtin_amdgcn_wmma_f32_16x16x32_bf16(
            false, a[m], false, b[n], (short)0, c[m][n], false, false);
      }
    }
  }

  // Epilogue. C/D layout: VGPR r -> M = r + 8*(lane>=16); N = n*16 + (lane&15).
  const int laneHi = lane >> 4;
  const int nl     = lane & 15;
  const int nColBase = nB0 * 16;               // first column of this wave
  const int chunk    = nB0 >> 2;               // 64-col chunk id (0..79)

#pragma unroll
  for (int m = 0; m < 4; ++m) {
    float rowAcc[8];
#pragma unroll
    for (int r = 0; r < 8; ++r) rowAcc[r] = 0.0f;
#pragma unroll
    for (int r = 0; r < 8; ++r) {
      const size_t row = (size_t)((mB0 + m) * 16 + r + (laneHi << 3));
      const size_t rb  = row * LC;
#pragma unroll
      for (int n = 0; n < 4; ++n) {
        rowAcc[r] += c[m][n][r] * x[rb + nColBase + n * 16 + nl];
      }
    }
#pragma unroll
    for (int r = 0; r < 8; ++r) {
      float v = rowAcc[r];
      v += __shfl_xor(v, 1, 32);
      v += __shfl_xor(v, 2, 32);
      v += __shfl_xor(v, 4, 32);
      v += __shfl_xor(v, 8, 32);   // reduce over the 16 lanes sharing this row
      if (nl == 0) {
        int row = (mB0 + m) * 16 + r + (laneHi << 3);
        partial[(size_t)row * NCHUNKS + chunk] = v;
      }
    }
  }
}

// ---------------------------------------------------------------------------
// Finalize: out[b] = theta0 + dot(x[b], theta_lc) + sum_chunks partial[b][*]
// Deterministic fixed-tree block reduction.
// ---------------------------------------------------------------------------
__global__ void __launch_bounds__(256) pgm_finalize(
    const float* __restrict__ x,
    const float* __restrict__ theta0,
    const float* __restrict__ theta_lc,
    const float* __restrict__ partial,
    float* __restrict__ out) {
  const int b   = blockIdx.x;
  const int tid = threadIdx.x;
  float s = 0.0f;
  const float* xr = x + (size_t)b * LC;
  for (int j = tid; j < LC; j += 256) s += xr[j] * theta_lc[j];
  for (int p = tid; p < NCHUNKS; p += 256) s += partial[(size_t)b * NCHUNKS + p];
  __shared__ float red[256];
  red[tid] = s;
  __syncthreads();
#pragma unroll
  for (int stride = 128; stride > 0; stride >>= 1) {
    if (tid < stride) red[tid] += red[tid + stride];
    __syncthreads();
  }
  if (tid == 0) out[b] = theta0[0] + red[0];
}

extern "C" void kernel_launch(void* const* d_in, const int* in_sizes, int n_in,
                              void* d_out, int out_size, void* d_ws, size_t ws_size,
                              hipStream_t stream) {
  (void)in_sizes; (void)n_in; (void)out_size; (void)ws_size;
  const float* x_lc       = (const float*)d_in[0];   // (B, L*C)
  const float* theta_0    = (const float*)d_in[1];   // (1,)
  const float* theta_lc   = (const float*)d_in[2];   // (1, L, C)
  const float* theta_lclc = (const float*)d_in[3];   // (1,L,C,L,C) — already masked
  // d_in[4] = mask: unused (theta_lclc already has it applied)

  // Workspace layout
  const size_t THETA_FRAG_ELEMS = (size_t)KSTEPS * NBLK * 512;  // 26,214,400 bf16
  const size_t X_FRAG_ELEMS     = (size_t)MBLK * KSTEPS * 512;  // 20,971,520 bf16
  unsigned short* thetaB = (unsigned short*)d_ws;
  unsigned short* aFrag  = thetaB + THETA_FRAG_ELEMS;
  float*          part   = (float*)(aFrag + X_FRAG_ELEMS);      // B * NCHUNKS f32
  float*          out    = (float*)d_out;

  {
    size_t n = (size_t)LC * LC;
    pgm_swizzle_theta<<<dim3((unsigned)((n + 255) / 256)), dim3(256), 0, stream>>>(
        theta_lclc, thetaB);
  }
  {
    size_t n = (size_t)BB * LC;
    pgm_swizzle_x<<<dim3((unsigned)((n + 255) / 256)), dim3(256), 0, stream>>>(
        x_lc, aFrag);
  }
  // Block = 8 waves (4 in M x 2 in N); wave tile 64x64; block tile 256x128.
  pgm_gemm<<<dim3(LC / 128, BB / 256), dim3(256), 0, stream>>>(aFrag, thetaB, x_lc, part);
  pgm_finalize<<<dim3(BB), dim3(256), 0, stream>>>(x_lc, theta_0, theta_lc, part, out);
}
